// MolGraphAutoencoder_60902636257736
// MI455X (gfx1250) — matrix operand order repaired and checked by reading, not compiled, not run on hardware
//
#include <hip/hip_runtime.h>

typedef _Float16 h16;
typedef __attribute__((ext_vector_type(8)))  _Float16 v8h;
typedef __attribute__((ext_vector_type(16))) _Float16 v16h;
typedef __attribute__((ext_vector_type(8)))  float    v8f;

#define NN 50000
#define EE 200000
#define GG 2048

static constexpr size_t AL(size_t x) { return (x + 255) & ~size_t(255); }

// ---- workspace layout (bytes) ----
static constexpr size_t o_xh     = 0;                                     // [N,64]  f16
static constexpr size_t o_agg0h  = AL(o_xh     + (size_t)NN*64*2);        // [N,64]  f16
static constexpr size_t o_h2h    = AL(o_agg0h  + (size_t)NN*64*2);        // [N,128] f16
static constexpr size_t o_agg2h  = AL(o_h2h    + (size_t)NN*128*2);       // [N,128] f16
static constexpr size_t o_h1     = AL(o_agg2h  + (size_t)NN*128*2);       // [N,1024] f16 (K-padded)
static constexpr size_t o_h3     = AL(o_h1     + (size_t)NN*1024*2);      // [N,1024] f16
static constexpr size_t o_w1rel  = AL(o_h3     + (size_t)NN*1024*2);      // [1008,64] f16
static constexpr size_t o_w1root = AL(o_w1rel  + (size_t)1008*64*2);
static constexpr size_t o_w2rel  = AL(o_w1root + (size_t)1008*64*2);      // [128,1024] f16
static constexpr size_t o_w2root = AL(o_w2rel  + (size_t)128*1024*2);
static constexpr size_t o_w3rel  = AL(o_w2root + (size_t)128*1024*2);     // [1008,128] f16
static constexpr size_t o_w3root = AL(o_w3rel  + (size_t)1008*128*2);
static constexpr size_t o_w4rel  = AL(o_w3root + (size_t)1008*128*2);     // [64,1024] f16
static constexpr size_t o_w4root = AL(o_w4rel  + (size_t)64*1024*2);
static constexpr size_t o_agg0   = AL(o_w4root + (size_t)64*1024*2);      // [N,64]  f32
static constexpr size_t o_y2     = AL(o_agg0   + (size_t)NN*64*4);        // [N,128] f32
static constexpr size_t o_r2     = AL(o_y2     + (size_t)NN*128*4);
static constexpr size_t o_aggy2  = AL(o_r2     + (size_t)NN*128*4);
static constexpr size_t o_h2f    = AL(o_aggy2  + (size_t)NN*128*4);
static constexpr size_t o_agg2   = AL(o_h2f    + (size_t)NN*128*4);
static constexpr size_t o_y4     = AL(o_agg2   + (size_t)NN*128*4);       // [N,64] f32
static constexpr size_t o_r4     = AL(o_y4     + (size_t)NN*64*4);
static constexpr size_t o_aggy4  = AL(o_r4     + (size_t)NN*64*4);
static constexpr size_t o_encsum = AL(o_aggy4  + (size_t)NN*64*4);        // [G,128] f32
static constexpr size_t o_enccnt = AL(o_encsum + (size_t)GG*128*4);       // [G] f32

// ---------------- elementwise helpers ----------------
__global__ void k_f32_to_f16(const float* __restrict__ in, h16* __restrict__ out, int n) {
  int i = blockIdx.x * 256 + threadIdx.x;
  if (i < n) out[i] = (h16)in[i];
}

// convert weight [dout,din] f32 -> padded [doutp,kp] f16 (zero pad)
__global__ void k_w_pad(const float* __restrict__ W, h16* __restrict__ Wh,
                        int dout, int din, int doutp, int kp) {
  int i = blockIdx.x * 256 + threadIdx.x;
  if (i >= doutp * kp) return;
  int n = i / kp, k = i - n * kp;
  float v = (n < dout && k < din) ? W[(size_t)n * din + k] : 0.0f;
  Wh[i] = (h16)v;
}

// agg[dst] += feat[src] over edges, feature dim = 1<<dl
__global__ void k_scatter_add(const float* __restrict__ feat, float* __restrict__ agg,
                              const int* __restrict__ ei, int dl) {
  long i = (long)blockIdx.x * 256 + threadIdx.x;
  long total = (long)EE << dl;
  if (i >= total) return;
  int d = 1 << dl;
  int e = (int)(i >> dl);
  int f = (int)(i & (d - 1));
  int s = ei[e], t = ei[EE + e];
  atomicAdd(&agg[(size_t)t * d + f], feat[(size_t)s * d + f]);
}

__global__ void k_h2(const float* __restrict__ aggy2, const float* __restrict__ r2,
                     float* __restrict__ h2f, h16* __restrict__ h2h) {
  int i = blockIdx.x * 256 + threadIdx.x;
  if (i >= NN * 128) return;
  float v = aggy2[i] + r2[i];
  v = v > 0.0f ? v : 0.0f;
  h2f[i] = v;
  h2h[i] = (h16)v;
}

__global__ void k_pool_sum(const float* __restrict__ h2, const int* __restrict__ batch,
                           float* __restrict__ encsum) {
  int i = blockIdx.x * 256 + threadIdx.x;
  if (i >= NN * 128) return;
  int n = i >> 7, f = i & 127;
  atomicAdd(&encsum[(size_t)batch[n] * 128 + f], h2[i]);
}

__global__ void k_count(const int* __restrict__ batch, float* __restrict__ cnt) {
  int i = blockIdx.x * 256 + threadIdx.x;
  if (i < NN) atomicAdd(&cnt[batch[i]], 1.0f);
}

__global__ void k_pool_div(const float* __restrict__ encsum, const float* __restrict__ cnt,
                           float* __restrict__ outenc) {
  int i = blockIdx.x * 256 + threadIdx.x;
  if (i >= GG * 128) return;
  outenc[i] = encsum[i] / fmaxf(cnt[i >> 7], 1.0f);
}

__global__ void k_out(const float* __restrict__ aggy4, const float* __restrict__ r4,
                      float* __restrict__ out) {
  int i = blockIdx.x * 256 + threadIdx.x;
  if (i >= NN * 64) return;
  out[i] = aggy4[i] + r4[i];
}

// ---------------- WMMA GEMMs ----------------
// A-fragment loader: contiguous K order for V_WMMA_F32_16X16X32_F16 A operand.
// lane lo = M row, hi selects K-halves {0-15} vs {16-31}.
__device__ __forceinline__ v16h load_afrag(const h16* __restrict__ p) {
  v8h lo8 = *(const v8h*)(p);
  v8h hi8 = *(const v8h*)(p + 8);
  return __builtin_shufflevector(lo8, hi8, 0,1,2,3,4,5,6,7,8,9,10,11,12,13,14,15);
}

__device__ __forceinline__ v8f wmma_f16(v16h a, v16h b, v8f c) {
  return __builtin_amdgcn_wmma_f32_16x16x32_f16(false, a, false, b, (short)0, c, false, false);
}

// D[m,n] = relu?( A1@W1^T + A2@W2^T + bias ), f16 out, row stride Dstride.
// 2 M-tiles per wave (B-fragments reused); 8 waves/block over consecutive n-tiles.
// K-loop is specialized on the second-tile guard so each variant stays straight-line
// (keeps the software pipeliner's partial s_wait_loadcnt overlap).
template <bool RELU>
__global__ __launch_bounds__(256) void k_gemm_dualA(
    const h16* __restrict__ A1, const h16* __restrict__ A2,
    const h16* __restrict__ W1, const h16* __restrict__ W2,
    const float* __restrict__ bias, h16* __restrict__ D,
    int K, int Dstride, int dreal, int ntiles) {
  int wave = threadIdx.x >> 5;
  int lane = threadIdx.x & 31;
  int ntile = blockIdx.y * 8 + wave;
  if (ntile >= ntiles) return;                  // wave-uniform: EXEC stays all-ones
  int m0 = blockIdx.x * 32;
  bool two = (m0 + 32) <= NN;                   // wave-uniform second-tile guard
  int n0 = ntile * 16;
  int hi = lane >> 4, lo = lane & 15;

  const h16* a1p0 = A1 + (size_t)(m0 + lo) * K + hi * 16;
  const h16* a2p0 = A2 + (size_t)(m0 + lo) * K + hi * 16;
  const h16* a1p1 = a1p0 + (size_t)16 * K;
  const h16* a2p1 = a2p0 + (size_t)16 * K;
  const h16* b1p  = W1 + (size_t)(n0 + lo) * K + hi * 16;
  const h16* b2p  = W2 + (size_t)(n0 + lo) * K + hi * 16;

  v8f acc0 = {};
  v8f acc1 = {};
  if (two) {
    for (int kk = 0; kk < K; kk += 32) {
      v16h b1 = *(const v16h*)(b1p + kk);
      v16h b2 = *(const v16h*)(b2p + kk);
      v16h a10 = load_afrag(a1p0 + kk);
      v16h a20 = load_afrag(a2p0 + kk);
      v16h a11 = load_afrag(a1p1 + kk);
      v16h a21 = load_afrag(a2p1 + kk);
      acc0 = wmma_f16(a10, b1, acc0);
      acc0 = wmma_f16(a20, b2, acc0);
      acc1 = wmma_f16(a11, b1, acc1);
      acc1 = wmma_f16(a21, b2, acc1);
    }
  } else {
    for (int kk = 0; kk < K; kk += 32) {
      v16h b1 = *(const v16h*)(b1p + kk);
      v16h b2 = *(const v16h*)(b2p + kk);
      v16h a10 = load_afrag(a1p0 + kk);
      v16h a20 = load_afrag(a2p0 + kk);
      acc0 = wmma_f16(a10, b1, acc0);
      acc0 = wmma_f16(a20, b2, acc0);
    }
  }
  int col = n0 + lo;
  float bv = (col < dreal) ? bias[col] : 0.0f;
  h16* dp0 = D + (size_t)(m0 + hi * 8) * Dstride + col;
#pragma unroll
  for (int g = 0; g < 8; ++g) {
    float v = acc0[g] + bv;
    if (RELU) v = v > 0.0f ? v : 0.0f;
    dp0[(size_t)g * Dstride] = (h16)v;
  }
  if (two) {
    h16* dp1 = D + (size_t)(m0 + 16 + hi * 8) * Dstride + col;
#pragma unroll
    for (int g = 0; g < 8; ++g) {
      float v = acc1[g] + bv;
      if (RELU) v = v > 0.0f ? v : 0.0f;
      dp1[(size_t)g * Dstride] = (h16)v;
    }
  }
}

// D1 = A@W1^T (no bias), D2 = A@W2^T + bias2; f32 outputs with row stride dout.
// 2 M-tiles per wave; one speculative prefetch of the next M-block before the loop.
__global__ __launch_bounds__(256) void k_gemm_dualW(
    const h16* __restrict__ A,
    const h16* __restrict__ W1, const h16* __restrict__ W2,
    const float* __restrict__ bias2,
    float* __restrict__ D1, float* __restrict__ D2,
    int K, int dout, int ntiles) {
  int wave = threadIdx.x >> 5;
  int lane = threadIdx.x & 31;
  int ntile = blockIdx.y * 8 + wave;
  if (ntile >= ntiles) return;
  int m0 = blockIdx.x * 32;
  bool two = (m0 + 32) <= NN;
  int n0 = ntile * 16;
  int hi = lane >> 4, lo = lane & 15;

  const h16* ap0 = A + (size_t)(m0 + lo) * K + hi * 16;
  const h16* ap1 = ap0 + (size_t)16 * K;
  const h16* b1p = W1 + (size_t)(n0 + lo) * K + hi * 16;
  const h16* b2p = W2 + (size_t)(n0 + lo) * K + hi * 16;

  // speculative warm-up of the next M-block of the streaming A operand
  __builtin_prefetch(ap0 + (size_t)32 * K, 0, 1);

  v8f accA0 = {};
  v8f accB0 = {};
  v8f accA1 = {};
  v8f accB1 = {};
  if (two) {
    for (int kk = 0; kk < K; kk += 32) {
      v16h b1 = *(const v16h*)(b1p + kk);
      v16h b2 = *(const v16h*)(b2p + kk);
      v16h a0 = load_afrag(ap0 + kk);
      v16h a1 = load_afrag(ap1 + kk);
      accA0 = wmma_f16(a0, b1, accA0);
      accB0 = wmma_f16(a0, b2, accB0);
      accA1 = wmma_f16(a1, b1, accA1);
      accB1 = wmma_f16(a1, b2, accB1);
    }
  } else {
    for (int kk = 0; kk < K; kk += 32) {
      v16h b1 = *(const v16h*)(b1p + kk);
      v16h b2 = *(const v16h*)(b2p + kk);
      v16h a0 = load_afrag(ap0 + kk);
      accA0 = wmma_f16(a0, b1, accA0);
      accB0 = wmma_f16(a0, b2, accB0);
    }
  }
  int col = n0 + lo;
  float bv = bias2[col];
  float* d1 = D1 + (size_t)(m0 + hi * 8) * dout + col;
  float* d2 = D2 + (size_t)(m0 + hi * 8) * dout + col;
#pragma unroll
  for (int g = 0; g < 8; ++g) {
    d1[(size_t)g * dout] = accA0[g];
    d2[(size_t)g * dout] = accB0[g] + bv;
  }
  if (two) {
    float* d3 = D1 + (size_t)(m0 + 16 + hi * 8) * dout + col;
    float* d4 = D2 + (size_t)(m0 + 16 + hi * 8) * dout + col;
#pragma unroll
    for (int g = 0; g < 8; ++g) {
      d3[(size_t)g * dout] = accA1[g];
      d4[(size_t)g * dout] = accB1[g] + bv;
    }
  }
}

// ---------------- launcher ----------------
extern "C" void kernel_launch(void* const* d_in, const int* in_sizes, int n_in,
                              void* d_out, int out_size, void* d_ws, size_t ws_size,
                              hipStream_t stream) {
  const float* x      = (const float*)d_in[0];
  const int*   ei     = (const int*)d_in[1];
  const int*   batch  = (const int*)d_in[2];
  const float* W1rel  = (const float*)d_in[3];
  const float* b1     = (const float*)d_in[4];
  const float* W1root = (const float*)d_in[5];
  const float* W2rel  = (const float*)d_in[6];
  const float* b2     = (const float*)d_in[7];
  const float* W2root = (const float*)d_in[8];
  const float* W3rel  = (const float*)d_in[9];
  const float* b3     = (const float*)d_in[10];
  const float* W3root = (const float*)d_in[11];
  const float* W4rel  = (const float*)d_in[12];
  const float* b4     = (const float*)d_in[13];
  const float* W4root = (const float*)d_in[14];
  float* out = (float*)d_out;

  char* ws = (char*)d_ws;
  auto H = [&](size_t o) { return (h16*)(ws + o); };
  auto F = [&](size_t o) { return (float*)(ws + o); };

  // zero: aggregation buffers, pooling buffers, padded h1/h3
  hipMemsetAsync(F(o_agg0),  0, (size_t)NN * 64 * 4,  stream);
  hipMemsetAsync(F(o_aggy2), 0, (size_t)NN * 128 * 4, stream);
  hipMemsetAsync(F(o_agg2),  0, (size_t)NN * 128 * 4, stream);
  hipMemsetAsync(F(o_aggy4), 0, (size_t)NN * 64 * 4,  stream);
  hipMemsetAsync(F(o_encsum), 0, (size_t)GG * 128 * 4, stream);
  hipMemsetAsync(F(o_enccnt), 0, (size_t)GG * 4,       stream);
  hipMemsetAsync(H(o_h1), 0, (size_t)NN * 1024 * 2, stream);
  hipMemsetAsync(H(o_h3), 0, (size_t)NN * 1024 * 2, stream);

  auto CEIL = [](long a, long b) { return (int)((a + b - 1) / b); };
  const int MB = CEIL(NN, 32);   // 1563 M-blocks of 2x16 rows

  // weight conversion + padding to f16
  k_w_pad<<<CEIL(1008 * 64, 256),  256, 0, stream>>>(W1rel,  H(o_w1rel),  1000, 64,   1008, 64);
  k_w_pad<<<CEIL(1008 * 64, 256),  256, 0, stream>>>(W1root, H(o_w1root), 1000, 64,   1008, 64);
  k_w_pad<<<CEIL(128 * 1024, 256), 256, 0, stream>>>(W2rel,  H(o_w2rel),  128,  1000, 128,  1024);
  k_w_pad<<<CEIL(128 * 1024, 256), 256, 0, stream>>>(W2root, H(o_w2root), 128,  1000, 128,  1024);
  k_w_pad<<<CEIL(1008 * 128, 256), 256, 0, stream>>>(W3rel,  H(o_w3rel),  1000, 128,  1008, 128);
  k_w_pad<<<CEIL(1008 * 128, 256), 256, 0, stream>>>(W3root, H(o_w3root), 1000, 128,  1008, 128);
  k_w_pad<<<CEIL(64 * 1024, 256),  256, 0, stream>>>(W4rel,  H(o_w4rel),  64,   1000, 64,   1024);
  k_w_pad<<<CEIL(64 * 1024, 256),  256, 0, stream>>>(W4root, H(o_w4root), 64,   1000, 64,   1024);

  // x -> f16
  k_f32_to_f16<<<CEIL((long)NN * 64, 256), 256, 0, stream>>>(x, H(o_xh), NN * 64);

  // L1: agg0 = scatter(x) [d=64]; h1 = relu(agg0@W1rel^T + x@W1root^T + b1)
  k_scatter_add<<<CEIL((long)EE * 64, 256), 256, 0, stream>>>(x, F(o_agg0), ei, 6);
  k_f32_to_f16<<<CEIL((long)NN * 64, 256), 256, 0, stream>>>(F(o_agg0), H(o_agg0h), NN * 64);
  k_gemm_dualA<true><<<dim3(MB, 8), 256, 0, stream>>>(
      H(o_agg0h), H(o_xh), H(o_w1rel), H(o_w1root), b1, H(o_h1), 64, 1024, 1000, 63);

  // L2: y2 = h1@W2rel^T ; r2 = h1@W2root^T + b2 ; h2 = relu(scatter(y2)+r2)
  k_gemm_dualW<<<dim3(MB, 1), 256, 0, stream>>>(
      H(o_h1), H(o_w2rel), H(o_w2root), b2, F(o_y2), F(o_r2), 1024, 128, 8);
  k_scatter_add<<<CEIL((long)EE * 128, 256), 256, 0, stream>>>(F(o_y2), F(o_aggy2), ei, 7);
  k_h2<<<CEIL((long)NN * 128, 256), 256, 0, stream>>>(F(o_aggy2), F(o_r2), F(o_h2f), H(o_h2h));

  // pooling -> encoded part of output
  k_pool_sum<<<CEIL((long)NN * 128, 256), 256, 0, stream>>>(F(o_h2f), batch, F(o_encsum));
  k_count<<<CEIL((long)NN, 256), 256, 0, stream>>>(batch, F(o_enccnt));
  k_pool_div<<<CEIL((long)GG * 128, 256), 256, 0, stream>>>(
      F(o_encsum), F(o_enccnt), out + (size_t)NN * 64);

  // L3: agg2 = scatter(h2) [d=128]; h3 = relu(agg2@W3rel^T + h2@W3root^T + b3)
  k_scatter_add<<<CEIL((long)EE * 128, 256), 256, 0, stream>>>(F(o_h2f), F(o_agg2), ei, 7);
  k_f32_to_f16<<<CEIL((long)NN * 128, 256), 256, 0, stream>>>(F(o_agg2), H(o_agg2h), NN * 128);
  k_gemm_dualA<true><<<dim3(MB, 8), 256, 0, stream>>>(
      H(o_agg2h), H(o_h2h), H(o_w3rel), H(o_w3root), b3, H(o_h3), 128, 1024, 1000, 63);

  // L4: y4 = h3@W4rel^T ; r4 = h3@W4root^T + b4 ; out = scatter(y4)+r4
  k_gemm_dualW<<<dim3(MB, 1), 256, 0, stream>>>(
      H(o_h3), H(o_w4rel), H(o_w4root), b4, F(o_y4), F(o_r4), 1024, 64, 4);
  k_scatter_add<<<CEIL((long)EE * 64, 256), 256, 0, stream>>>(F(o_y4), F(o_aggy4), ei, 6);
  k_out<<<CEIL((long)NN * 64, 256), 256, 0, stream>>>(F(o_aggy4), F(o_r4), out);
}
